// lora_cross_attention_ttoi_1941325218361
// MI455X (gfx1250) — matrix operand order repaired and checked by reading, not compiled, not run on hardware
//
#include <hip/hip_runtime.h>

#define DEV __device__ __forceinline__

typedef __attribute__((ext_vector_type(16))) __bf16         v16bf;
typedef __attribute__((ext_vector_type(8)))  float          v8f;
typedef __attribute__((ext_vector_type(4)))  unsigned int   v4u;
typedef __attribute__((ext_vector_type(8)))  unsigned int   v8u;
typedef __attribute__((ext_vector_type(4)))  float          v4f;
typedef __attribute__((ext_vector_type(4)))  unsigned short v4h;

union F16B { v4u u[2]; v16bf v; };

DEV unsigned short f32_to_bf16(float f) {
  unsigned int u = __float_as_uint(f);
  u += 0x7FFFu + ((u >> 16) & 1u);   // round-to-nearest-even
  return (unsigned short)(u >> 16);
}

// A-fragment (16x32 bf16): lane L: M=L%16, halves hold K = {8h..8h+7, 8h+16..8h+23}
DEV v16bf load_afrag(const unsigned short* p /* row base + kcol + 8*half */) {
  F16B f;
  f.u[0] = *(const v4u*)p;
  f.u[1] = *(const v4u*)(p + 16);
  return f.v;
}
// B-fragment (32x16 bf16): lane L: N=L%16, halves hold K = {8h..8h+7, 8h+16..8h+23}
DEV v16bf load_bfrag(const unsigned short* p /* row base + kcol + 8*half */) {
  F16B f;
  f.u[0] = *(const v4u*)p;
  f.u[1] = *(const v4u*)(p + 16);
  return f.v;
}

DEV float lane_xor_f(float v, int off) {
  int idx = (((int)(threadIdx.x & 31)) ^ off) << 2;
  return __int_as_float(__builtin_amdgcn_ds_bpermute(idx, __float_as_int(v)));
}

// ---------------------------------------------------------------- TDM (Tensor Data Mover)
// 2D tile load, data_size=4B elements, 2-SGPR-group descriptor (VADDR2/3 = NULL form).
// Tile: tile_w_dw dwords per row, tile_h rows; global row stride = stride_dw dwords.
// LDS padding: after every 16 dwords insert 4 dwords -> LDS row stride 40 ushorts (80B).
// tensor_dim1 = tensor_h: rows >= tensor_h read as zero (M-edge handling).
DEV void tdm_load_2d_16dw(unsigned int lds_addr, const void* gptr,
                          unsigned int tile_h, unsigned int stride_dw,
                          unsigned int tensor_h) {
  const unsigned int tile_w_dw = 16u;
  unsigned long long ga = (unsigned long long)gptr;
  v4u g0;
  g0[0] = 1u;                                        // count=1, user descriptor
  g0[1] = lds_addr;                                  // LDS byte address
  g0[2] = (unsigned int)ga;                          // global_addr[31:0]
  g0[3] = ((unsigned int)(ga >> 32) & 0x01FFFFFFu)   // global_addr[56:32]
          | 0x80000000u;                             // type=2 ("image") at bits 127:126
  v8u g1;
  g1[0] = (2u << 16)                                 // data_size: 4 bytes
        | (1u << 20)                                 // pad_enable
        | (3u << 22)                                 // pad_interval: 16 dwords
        | (3u << 25);                                // pad_amount: 4 dwords
  g1[1] = (tile_w_dw & 0xFFFFu) << 16;               // tensor_dim0[15:0] (=tile width)
  g1[2] = ((tile_w_dw >> 16) & 0xFFFFu)              // tensor_dim0[31:16]
        | ((tensor_h & 0xFFFFu) << 16);              // tensor_dim1[15:0]
  g1[3] = ((tensor_h >> 16) & 0xFFFFu)               // tensor_dim1[31:16]
        | (tile_w_dw << 16);                         // tile_dim0
  g1[4] = tile_h & 0xFFFFu;                          // tile_dim1 (tile_dim2 = 0)
  g1[5] = stride_dw;                                 // tensor_dim0_stride[31:0]
  g1[6] = 0u;                                        // stride[47:32]=0, dim1_stride lo
  g1[7] = 0u;
  asm volatile("tensor_load_to_lds %0, %1" :: "s"(g0), "s"(g1) : "memory");
}

// ---------------------------------------------------------------- convert
__global__ __launch_bounds__(256)
void k_cvt_f32_bf16(const float* __restrict__ src, unsigned short* __restrict__ dst, int n4) {
  int i = blockIdx.x * blockDim.x + threadIdx.x;
  int stride = gridDim.x * blockDim.x;
  for (; i < n4; i += stride) {
    v4f x = *(const v4f*)(src + 4 * (size_t)i);
    v4h y = { f32_to_bf16(x[0]), f32_to_bf16(x[1]), f32_to_bf16(x[2]), f32_to_bf16(x[3]) };
    *(v4h*)(dst + 4 * (size_t)i) = y;
  }
}

// ---------------------------------------------------------------- GEMM  C = A * W^T + b
// A: [M,K] bf16 row-major, W: [N,K] bf16 row-major, bias f32 [N].
// Block tile 128x128, BK=32, 8 waves: 4(M) x 2(N), wave tile 32x64.
// TDM double-buffered LDS: wave 0 streams next tiles while all waves run WMMAs.
template<bool OUT_BF16>
__global__ __launch_bounds__(256)
void k_gemm_xWT(const unsigned short* __restrict__ A,
                const unsigned short* __restrict__ W,
                const float* __restrict__ bias,
                void* __restrict__ Cout,
                int M, int N, int K)
{
  constexpr int BM = 128, BN = 128, BK = 32;
  constexpr int LDT = 40;  // 32 data halves + 4-dword TDM pad: 80B row, 16B aligned, 20-bank rotation
  __shared__ __align__(16) unsigned short sA[2][BM * LDT];
  __shared__ __align__(16) unsigned short sB[2][BN * LDT];

  const int tid  = threadIdx.x;
  const int lane = tid & 31, wv = tid >> 5;
  const int hv   = lane >> 4, l16 = lane & 15;
  const int wM   = wv & 3, wN = wv >> 2;          // 4 x 2 wave grid
  const int m0   = blockIdx.y * BM;
  const int n0   = blockIdx.x * BN;

  v8f acc[2][4] = {};

  const int nIter = K / BK;                        // 48
  if (wv == 0) {
    tdm_load_2d_16dw((unsigned int)(size_t)&sA[0][0], A + (size_t)m0 * K, BM, K / 2, (unsigned)(M - m0));
    tdm_load_2d_16dw((unsigned int)(size_t)&sB[0][0], W + (size_t)n0 * K, BN, K / 2, (unsigned)(N - n0));
  }

  int p = 0;
  for (int it = 0; it < nIter; ++it) {
    __syncthreads();                               // everyone done reading buf[1-p]
    if (wv == 0) {
      if (it + 1 < nIter) {
        const int k1 = (it + 1) * BK;
        tdm_load_2d_16dw((unsigned int)(size_t)&sA[1 - p][0], A + (size_t)m0 * K + k1, BM, K / 2, (unsigned)(M - m0));
        tdm_load_2d_16dw((unsigned int)(size_t)&sB[1 - p][0], W + (size_t)n0 * K + k1, BN, K / 2, (unsigned)(N - n0));
        __builtin_amdgcn_s_wait_tensorcnt(2);      // pair for buf[p] retired; next pair in flight
      } else {
        __builtin_amdgcn_s_wait_tensorcnt(0);
      }
    }
    __syncthreads();                               // release buf[p] to all waves

    const unsigned short* cA = sA[p];
    const unsigned short* cB = sB[p];
    v16bf af[2], bfr[4];
#pragma unroll
    for (int mi = 0; mi < 2; ++mi)
      af[mi] = load_afrag(cA + (wM * 32 + mi * 16 + l16) * LDT + hv * 8);
#pragma unroll
    for (int ni = 0; ni < 4; ++ni)
      bfr[ni] = load_bfrag(cB + (wN * 64 + ni * 16 + l16) * LDT + hv * 8);
#pragma unroll
    for (int mi = 0; mi < 2; ++mi)
#pragma unroll
      for (int ni = 0; ni < 4; ++ni)
        acc[mi][ni] = __builtin_amdgcn_wmma_f32_16x16x32_bf16(
            false, af[mi], false, bfr[ni], (short)0, acc[mi][ni], false, false);
    p ^= 1;
  }

  // ---- epilogue: C layout: vgpr r -> row (r + 8*half), lane%16 -> col
#pragma unroll
  for (int ni = 0; ni < 4; ++ni) {
    const int n  = n0 + wN * 64 + ni * 16 + l16;
    const float bn = bias[n];
#pragma unroll
    for (int mi = 0; mi < 2; ++mi) {
      const int mbase = m0 + wM * 32 + mi * 16 + hv * 8;
#pragma unroll
      for (int r = 0; r < 8; ++r) {
        const int m = mbase + r;
        if (m < M) {
          const float vv = acc[mi][ni][r] + bn;
          if (OUT_BF16)
            ((unsigned short*)Cout)[(size_t)m * N + n] = f32_to_bf16(vv);
          else
            ((float*)Cout)[(size_t)m * N + n] = vv;
        }
      }
    }
  }
}

// ---------------------------------------------------------------- attention
// 1 block = 4 waves = 64 q rows for one (b,h). Sk=77 (pad 80 for scores, 96 for PV), HD=192.
__global__ __launch_bounds__(128)
void k_attn(const unsigned short* __restrict__ Qb,   // [32,1024,1536] bf16
            const unsigned short* __restrict__ Kb,   // [32,77,1536]   bf16
            const unsigned short* __restrict__ Vb,   // [32,77,1536]   bf16
            unsigned short* __restrict__ Ob)         // [32,1024,1536] bf16
{
  constexpr int HD = 192, SK = 77, D = 1536;
  constexpr int VSTR = 104, PSTR = 104;              // 208B strides: 16B aligned, 4-bank rotation
  __shared__ __align__(16) unsigned short sVT[HD * VSTR];     // V^T: [d][j], j zero-padded to 96
  __shared__ __align__(16) unsigned short sP[4 * 16 * PSTR];  // per-wave P: [16][96] bf16

  const int tid  = threadIdx.x;
  const int lane = tid & 31, wv = tid >> 5;
  const int hv   = lane >> 4, l16 = lane & 15;
  const int b = blockIdx.z, h = blockIdx.y;
  const int q0 = blockIdx.x * 64 + wv * 16;

  // ---- stage V^T (coalesced global reads, strided LDS writes), zero pad j in [77,96)
  const unsigned short* Vh = Vb + (size_t)b * SK * D + h * HD;
  for (int idx = tid; idx < 96 * HD; idx += 128) {
    int j = idx / HD, d = idx - j * HD;
    unsigned short val = (j < SK) ? Vh[(size_t)j * D + d] : (unsigned short)0;
    sVT[d * VSTR + j] = val;
  }
  __syncthreads();

  // ---- Q fragments, straight from global (A layout: lane = q row, contiguous d)
  const unsigned short* Qrow = Qb + ((size_t)(b * 1024 + q0 + l16)) * D + h * HD;
  v16bf qf[6];
#pragma unroll
  for (int ks = 0; ks < 6; ++ks)
    qf[ks] = load_afrag(Qrow + ks * 32 + hv * 8);

  // ---- scores: 5 n-tiles x 6 k-steps of WMMA (B from global: lane = key row, contiguous d)
  const unsigned short* Kh = Kb + (size_t)b * SK * D + h * HD;
  v8f s[5];
#pragma unroll
  for (int t = 0; t < 5; ++t) {
    int n = t * 16 + l16; if (n > SK - 1) n = SK - 1;   // clamp (masked later)
    const unsigned short* Krow = Kh + (size_t)n * D;
    v8f a = {};
#pragma unroll
    for (int ks = 0; ks < 6; ++ks)
      a = __builtin_amdgcn_wmma_f32_16x16x32_bf16(
          false, qf[ks], false, load_bfrag(Krow + ks * 32 + hv * 8), (short)0, a, false, false);
    s[t] = a;
  }

  // ---- softmax over 80 cols (C layout: row = r + 8*half, col = t*16 + l16)
  const float scale = 0.07216878364870322f;  // 192^-0.5
#pragma unroll
  for (int t = 0; t < 5; ++t) {
    const bool valid = (t * 16 + l16) < SK;
#pragma unroll
    for (int r = 0; r < 8; ++r)
      s[t][r] = valid ? s[t][r] * scale : -1e30f;
  }
  float mx[8], sum[8];
#pragma unroll
  for (int r = 0; r < 8; ++r) {
    float m = s[0][r];
#pragma unroll
    for (int t = 1; t < 5; ++t) m = fmaxf(m, s[t][r]);
    mx[r] = m;
  }
#pragma unroll
  for (int off = 1; off < 16; off <<= 1)
#pragma unroll
    for (int r = 0; r < 8; ++r) mx[r] = fmaxf(mx[r], lane_xor_f(mx[r], off));
#pragma unroll
  for (int r = 0; r < 8; ++r) sum[r] = 0.f;
#pragma unroll
  for (int t = 0; t < 5; ++t)
#pragma unroll
    for (int r = 0; r < 8; ++r) {
      float e = __expf(s[t][r] - mx[r]);
      s[t][r] = e;
      sum[r] += e;
    }
#pragma unroll
  for (int off = 1; off < 16; off <<= 1)
#pragma unroll
    for (int r = 0; r < 8; ++r) sum[r] += lane_xor_f(sum[r], off);

  // ---- write normalized P to LDS as bf16, zero pad cols [80,96)
  unsigned short* Pw = sP + wv * 16 * PSTR;
#pragma unroll
  for (int t = 0; t < 5; ++t) {
#pragma unroll
    for (int r = 0; r < 8; ++r)
      Pw[(r + 8 * hv) * PSTR + t * 16 + l16] = f32_to_bf16(s[t][r] * (1.0f / sum[r]));
  }
#pragma unroll
  for (int r = 0; r < 8; ++r)
    Pw[(r + 8 * hv) * PSTR + 80 + l16] = 0;
  __syncthreads();   // cross-lane LDS visibility before fragment reads

  // ---- PV: K=96 (3 k-steps), 12 d-tiles
  v16bf pf[3];
#pragma unroll
  for (int ks = 0; ks < 3; ++ks)
    pf[ks] = load_afrag(Pw + l16 * PSTR + ks * 32 + hv * 8);

#pragma unroll
  for (int dt = 0; dt < 12; ++dt) {
    v8f o = {};
#pragma unroll
    for (int ks = 0; ks < 3; ++ks)
      o = __builtin_amdgcn_wmma_f32_16x16x32_bf16(
          false, pf[ks],
          false, load_bfrag(sVT + (dt * 16 + l16) * VSTR + ks * 32 + hv * 8),
          (short)0, o, false, false);
    const int d = h * HD + dt * 16 + l16;
#pragma unroll
    for (int r = 0; r < 8; ++r) {
      const int m = q0 + r + 8 * hv;
      Ob[((size_t)(b * 1024) + m) * D + d] = f32_to_bf16(o[r]);
    }
  }
}

// ---------------------------------------------------------------- launch
extern "C" void kernel_launch(void* const* d_in, const int* in_sizes, int n_in,
                              void* d_out, int out_size, void* d_ws, size_t ws_size,
                              hipStream_t stream) {
  (void)in_sizes; (void)n_in; (void)out_size; (void)ws_size;
  const float* img  = (const float*)d_in[0];
  const float* text = (const float*)d_in[1];
  const float* Wq   = (const float*)d_in[2];
  const float* bq   = (const float*)d_in[3];
  const float* Wk   = (const float*)d_in[4];
  const float* bk   = (const float*)d_in[5];
  const float* Wv   = (const float*)d_in[6];
  const float* bv   = (const float*)d_in[7];
  const float* Wo   = (const float*)d_in[8];
  const float* bo   = (const float*)d_in[9];

  constexpr size_t SZ_IMGB = (size_t)32 * 1024 * 1536 * 2;  // 100,663,296
  constexpr size_t SZ_TXTB = (size_t)32 * 77 * 1536 * 2;    //   7,569,408
  constexpr size_t SZ_W    = (size_t)1536 * 1536 * 2;       //   4,718,592

  char* ws = (char*)d_ws;
  unsigned short* imgb  = (unsigned short*)(ws + 0);                       // also reused for attn out
  unsigned short* attnb = imgb;                                            // alias: img dead after Q gemm
  unsigned short* txtb  = (unsigned short*)(ws + SZ_IMGB);
  unsigned short* wqb   = (unsigned short*)(ws + SZ_IMGB + SZ_TXTB);
  unsigned short* wkb   = (unsigned short*)(ws + SZ_IMGB + SZ_TXTB + 1 * SZ_W);
  unsigned short* wvb   = (unsigned short*)(ws + SZ_IMGB + SZ_TXTB + 2 * SZ_W);
  unsigned short* wob   = (unsigned short*)(ws + SZ_IMGB + SZ_TXTB + 3 * SZ_W);
  unsigned short* qb    = (unsigned short*)(ws + SZ_IMGB + SZ_TXTB + 4 * SZ_W);
  unsigned short* kb    = (unsigned short*)(ws + 2 * SZ_IMGB + SZ_TXTB + 4 * SZ_W);
  unsigned short* vb    = (unsigned short*)(ws + 2 * SZ_IMGB + 2 * SZ_TXTB + 4 * SZ_W);

  // 1) fp32 -> bf16 conversions
  k_cvt_f32_bf16<<<4096, 256, 0, stream>>>(img,  imgb, 32 * 1024 * 1536 / 4);
  k_cvt_f32_bf16<<<1024, 256, 0, stream>>>(text, txtb, 32 * 77 * 1536 / 4);
  k_cvt_f32_bf16<<<1024, 256, 0, stream>>>(Wq, wqb, 1536 * 1536 / 4);
  k_cvt_f32_bf16<<<1024, 256, 0, stream>>>(Wk, wkb, 1536 * 1536 / 4);
  k_cvt_f32_bf16<<<1024, 256, 0, stream>>>(Wv, wvb, 1536 * 1536 / 4);
  k_cvt_f32_bf16<<<1024, 256, 0, stream>>>(Wo, wob, 1536 * 1536 / 4);

  // 2) projections (bf16 out)
  k_gemm_xWT<true><<<dim3(12, 256), 256, 0, stream>>>(imgb, wqb, bq, qb, 32768, 1536, 1536);
  k_gemm_xWT<true><<<dim3(12, 20),  256, 0, stream>>>(txtb, wkb, bk, kb, 2464,  1536, 1536);
  k_gemm_xWT<true><<<dim3(12, 20),  256, 0, stream>>>(txtb, wvb, bv, vb, 2464,  1536, 1536);

  // 3) attention (writes bf16, aliases img buffer)
  k_attn<<<dim3(16, 8, 32), 128, 0, stream>>>(qb, kb, vb, attnb);

  // 4) output projection (f32 out -> d_out)
  k_gemm_xWT<false><<<dim3(12, 256), 256, 0, stream>>>(attnb, wob, bo, (float*)d_out, 32768, 1536, 1536);
}
